// MultiHeadAttention_5196910428832
// MI455X (gfx1250) — matrix operand order repaired and checked by reading, not compiled, not run on hardware
//
#include <hip/hip_runtime.h>
#include <hip/hip_bf16.h>
#include <math.h>

// ---------------------------------------------------------------------------
// MI455X (gfx1250) multi-head attention, bf16 WMMA everywhere.
// B=2, S=2048, D=1024, H=16, DK=64.
// ---------------------------------------------------------------------------

#define S_LEN   2048
#define DMODEL  1024
#define NHEADS  16
#define DKEY    64
#define MROWS   (2 * S_LEN)          // 4096 token rows (batch folded)

typedef unsigned short u16;
typedef __attribute__((ext_vector_type(16))) __bf16 v16bf;
typedef __attribute__((ext_vector_type(8)))  __bf16 v8bf;
typedef __attribute__((ext_vector_type(8)))  float  v8f;

// f32 -> bf16 bits, round-to-nearest-even
static __device__ inline u16 f2bf(float f) {
  unsigned u = __float_as_uint(f);
  unsigned r = u + 0x7FFFu + ((u >> 16) & 1u);
  return (u16)(r >> 16);
}

static __device__ inline v8f wmma_bf16(v16bf a, v16bf b, v8f c) {
  return __builtin_amdgcn_wmma_f32_16x16x32_bf16(false, a, false, b,
                                                 (short)0, c, false, false);
}

// ---- DPP16 row-rotate reductions over a 16-lane half-wave (VALU only, ----
// ---- keeps the DS pipe free for WMMA fragment traffic).               ----
// row_ror:n has no invalid lanes, and rotations {8,4,2,1} fold the full
// 16-lane reduction into every lane.
static __device__ inline float rowmax16(float x) {
  x = fmaxf(x, __int_as_float(__builtin_amdgcn_update_dpp(
          0, __float_as_int(x), 0x128, 0xF, 0xF, false)));   // row_ror:8
  x = fmaxf(x, __int_as_float(__builtin_amdgcn_update_dpp(
          0, __float_as_int(x), 0x124, 0xF, 0xF, false)));   // row_ror:4
  x = fmaxf(x, __int_as_float(__builtin_amdgcn_update_dpp(
          0, __float_as_int(x), 0x122, 0xF, 0xF, false)));   // row_ror:2
  x = fmaxf(x, __int_as_float(__builtin_amdgcn_update_dpp(
          0, __float_as_int(x), 0x121, 0xF, 0xF, false)));   // row_ror:1
  return x;
}
static __device__ inline float rowsum16(float x) {
  x += __int_as_float(__builtin_amdgcn_update_dpp(
          0, __float_as_int(x), 0x128, 0xF, 0xF, false));
  x += __int_as_float(__builtin_amdgcn_update_dpp(
          0, __float_as_int(x), 0x124, 0xF, 0xF, false));
  x += __int_as_float(__builtin_amdgcn_update_dpp(
          0, __float_as_int(x), 0x122, 0xF, 0xF, false));
  x += __int_as_float(__builtin_amdgcn_update_dpp(
          0, __float_as_int(x), 0x121, 0xF, 0xF, false));
  return x;
}

// A-fragment (16x32 bf16): lane holds row (lane&15); halves at K offsets
// (lane>>4)*8 and 16+(lane>>4)*8. base points at [row0][k0].
static __device__ inline v16bf load_afrag(const u16* base, int rowStride) {
  int lane = threadIdx.x & 31;
  const u16* p = base + (size_t)(lane & 15) * rowStride + ((lane >> 4) << 3);
  v8bf lo = *(const v8bf*)(p);
  v8bf hi = *(const v8bf*)(p + 16);
  v16bf r;
#pragma unroll
  for (int i = 0; i < 8; ++i) { r[i] = lo[i]; r[i + 8] = hi[i]; }
  return r;
}

// B-fragment (32x16 bf16) sourced from a row-major matrix whose ROWS are the
// WMMA N dimension and whose columns are the K dimension (i.e. X @ M^T form):
// lane holds column n=(lane&15) as row n of the source; halves at K offsets
// (lane>>4)*8 and 16+(lane>>4)*8.
static __device__ inline v16bf load_bfrag(const u16* base, int rowStride) {
  int lane = threadIdx.x & 31;
  const u16* p = base + (size_t)(lane & 15) * rowStride + ((lane >> 4) << 3);
  v8bf lo = *(const v8bf*)(p);
  v8bf hi = *(const v8bf*)(p + 16);
  v16bf r;
#pragma unroll
  for (int i = 0; i < 8; ++i) { r[i] = lo[i]; r[i + 8] = hi[i]; }
  return r;
}

// ---------------------------------------------------------------------------
// f32 -> bf16 conversion (bandwidth-bound prologue)
// ---------------------------------------------------------------------------
__global__ void cvt_f32_bf16(const float* __restrict__ in,
                             u16* __restrict__ out, int n) {
  int i = blockIdx.x * blockDim.x + threadIdx.x;
  if (i < n) out[i] = f2bf(in[i]);
}

// ---------------------------------------------------------------------------
// GEMM: C[M,N] = A[M,K] @ W[N,K]^T + bias.  A,W bf16; out bf16 or f32
// selected at compile time (branchless epilogue).
// Block = 256 threads = 8 waves arranged 2(M)x4(N); wave tile 32x64;
// block tile 64(M) x 256(N). grid = (N/256, M/64).
// ---------------------------------------------------------------------------
template <bool F32OUT>
__global__ __launch_bounds__(256)
void gemm_xwt(const u16* __restrict__ A, const u16* __restrict__ W,
              const float* __restrict__ bias, void* __restrict__ outP,
              int M, int N, int Kd) {
  int wave = threadIdx.x >> 5;
  int lane = threadIdx.x & 31;
  int wm = wave >> 2;                 // 0..1
  int wn = wave & 3;                  // 0..3
  int mbase = blockIdx.y * 64 + wm * 32;
  int nbase = blockIdx.x * 256 + wn * 64;

  v8f acc[2][4] = {};

  for (int k0 = 0; k0 < Kd; k0 += 32) {
    v16bf af[2];
    af[0] = load_afrag(A + (size_t)mbase * Kd + k0, Kd);
    af[1] = load_afrag(A + (size_t)(mbase + 16) * Kd + k0, Kd);
    v16bf bf[4];
#pragma unroll
    for (int nt = 0; nt < 4; ++nt)
      bf[nt] = load_bfrag(W + (size_t)(nbase + nt * 16) * Kd + k0, Kd);
#pragma unroll
    for (int mt = 0; mt < 2; ++mt)
#pragma unroll
      for (int nt = 0; nt < 4; ++nt)
        acc[mt][nt] = wmma_bf16(af[mt], bf[nt], acc[mt][nt]);
  }

  // epilogue: C layout row = v + 8*(lane>>4), col = lane&15
#pragma unroll
  for (int mt = 0; mt < 2; ++mt) {
#pragma unroll
    for (int nt = 0; nt < 4; ++nt) {
#pragma unroll
      for (int v = 0; v < 8; ++v) {
        int row = mbase + mt * 16 + v + ((lane >> 4) << 3);
        int col = nbase + nt * 16 + (lane & 15);
        float r = acc[mt][nt][v] + bias[col];
        if (F32OUT) ((float*)outP)[(size_t)row * N + col] = r;
        else        ((u16*)outP)[(size_t)row * N + col] = f2bf(r);
      }
    }
  }
}

// ---------------------------------------------------------------------------
// Flash attention: streaming softmax over keys, 32 keys per step.
// Block = 256 threads = 8 waves; each wave owns 16 query rows (full 64-d head).
// K tile (32x64) is staged via GLOBAL_LOAD_ASYNC_TO_LDS_B128 (ASYNCcnt path,
// bypasses VGPRs); V^T tile (64x32) is transposed through VGPRs while the
// async copy is in flight. P round-trips through per-wave LDS to become an
// A-fragment. Softmax row reductions use DPP row_ror (VALU pipe).
// grid = (B*H, S/128).
// ---------------------------------------------------------------------------
__global__ __launch_bounds__(256)
void attn_flash(const u16* __restrict__ Q, const u16* __restrict__ K,
                const u16* __restrict__ V, u16* __restrict__ Ocomb) {
  __shared__ u16 Ktile[32 * 64];        // [keyLocal][d]
  __shared__ u16 VTt[64 * 32];          // [d][keyLocal]
  __shared__ u16 Ptile[8][16 * 32];     // per-wave P staging

  int bh = blockIdx.x;                  // 0..31
  int b  = bh >> 4;
  int h  = bh & 15;
  int wave = threadIdx.x >> 5;
  int lane = threadIdx.x & 31;
  int qbase = blockIdx.y * 128 + wave * 16;

  const u16* Qp = Q + (size_t)b * S_LEN * DMODEL + h * DKEY;
  const u16* Kp = K + (size_t)b * S_LEN * DMODEL + h * DKEY;
  const u16* Vp = V + (size_t)b * S_LEN * DMODEL + h * DKEY;

  // Q fragments are loop-invariant: two K-windows over d = 0..63
  v16bf qf0 = load_afrag(Qp + (size_t)qbase * DMODEL + 0,  DMODEL);
  v16bf qf1 = load_afrag(Qp + (size_t)qbase * DMODEL + 32, DMODEL);

  v8f o[4] = {};
  float mrow[8], lrow[8];
#pragma unroll
  for (int v = 0; v < 8; ++v) { mrow[v] = -1e30f; lrow[v] = 0.0f; }

  const float scale = 0.125f;           // 1/sqrt(64)

  // Per-thread staging coordinates (8 bf16 elems per thread per tile)
  int keyl = threadIdx.x >> 3;          // 0..31
  int dc   = (threadIdx.x & 7) << 3;    // 0,8,...,56
  // LDS byte offset of this thread's K-tile destination (ISA: flat-address
  // low 32 bits are the LDS offset)
  unsigned ldsKdst = (unsigned)(size_t)&Ktile[keyl * 64 + dc];

  for (int kt = 0; kt < S_LEN / 32; ++kt) {
    __syncthreads();
    {
      size_t srow = (size_t)(kt * 32 + keyl) * DMODEL + dc;
      // --- async copy of K tile straight into LDS (16B per lane) ---
      unsigned long long gK = (unsigned long long)(size_t)(Kp + srow);
      asm volatile("global_load_async_to_lds_b128 %0, %1, off"
                   :: "v"(ldsKdst), "v"(gK) : "memory");
      // --- V tile transpose through VGPRs, overlapped with the async copy ---
      uint4 vv = *(const uint4*)(Vp + srow);
      u16 tmp[8];
      *(uint4*)tmp = vv;
#pragma unroll
      for (int i = 0; i < 8; ++i) VTt[(dc + i) * 32 + keyl] = tmp[i];
      asm volatile("s_wait_asynccnt 0x0" ::: "memory");
    }
    __syncthreads();

    // ---- scores: S[16 x 32] = Q(16x64) . K(32keys x 64)^T, two N halves
    v8f s0 = {}, s1 = {};
    s0 = wmma_bf16(qf0, load_bfrag(&Ktile[0 * 64 + 0],  64), s0);
    s0 = wmma_bf16(qf1, load_bfrag(&Ktile[0 * 64 + 32], 64), s0);
    s1 = wmma_bf16(qf0, load_bfrag(&Ktile[16 * 64 + 0],  64), s1);
    s1 = wmma_bf16(qf1, load_bfrag(&Ktile[16 * 64 + 32], 64), s1);

    // ---- online softmax per row (row = v + 8*(lane>>4), cols across lanes)
#pragma unroll
    for (int v = 0; v < 8; ++v) {
      float a0 = s0[v] * scale;
      float a1 = s1[v] * scale;
      float mx = rowmax16(fmaxf(a0, a1));
      float mnew  = fmaxf(mrow[v], mx);
      float alpha = __expf(mrow[v] - mnew);
      float p0 = __expf(a0 - mnew);
      float p1 = __expf(a1 - mnew);
      float rs = rowsum16(p0 + p1);
      lrow[v] = lrow[v] * alpha + rs;
      mrow[v] = mnew;
#pragma unroll
      for (int dt = 0; dt < 4; ++dt) o[dt][v] *= alpha;

      int row = v + ((lane >> 4) << 3);
      int col = lane & 15;
      Ptile[wave][row * 32 + col]      = f2bf(p0);
      Ptile[wave][row * 32 + col + 16] = f2bf(p1);
    }

    // same-wave LDS RAW: DS ops stay in order within a wave
    v16bf pf = load_afrag(&Ptile[wave][0], 32);

    // ---- O(16x64) += P(16x32) . V(32x64); B frags read rows of V^T
#pragma unroll
    for (int dt = 0; dt < 4; ++dt) {
      v16bf vf = load_bfrag(&VTt[(dt * 16) * 32], 32);
      o[dt] = wmma_bf16(pf, vf, o[dt]);
    }
  }

  // ---- normalize and write combined [b, s, h*64 + d] as bf16
#pragma unroll
  for (int v = 0; v < 8; ++v) {
    float inv = 1.0f / lrow[v];
    int row = qbase + v + ((lane >> 4) << 3);
    int col = lane & 15;
    size_t outBase = ((size_t)(b * S_LEN + row)) * DMODEL + h * DKEY;
#pragma unroll
    for (int dt = 0; dt < 4; ++dt)
      Ocomb[outBase + dt * 16 + col] = f2bf(o[dt][v] * inv);
  }
}

// ---------------------------------------------------------------------------
// Host-side launch
// ---------------------------------------------------------------------------
extern "C" void kernel_launch(void* const* d_in, const int* in_sizes, int n_in,
                              void* d_out, int out_size, void* d_ws, size_t ws_size,
                              hipStream_t stream) {
  (void)in_sizes; (void)n_in; (void)out_size; (void)ws_size;

  const float* query = (const float*)d_in[0];
  const float* key   = (const float*)d_in[1];
  const float* value = (const float*)d_in[2];
  const float* Wq    = (const float*)d_in[3];
  const float* bq    = (const float*)d_in[4];
  const float* Wk    = (const float*)d_in[5];
  const float* bk    = (const float*)d_in[6];
  const float* Wv    = (const float*)d_in[7];
  const float* bv    = (const float*)d_in[8];
  const float* Wo    = (const float*)d_in[9];
  const float* bo    = (const float*)d_in[10];
  float* out = (float*)d_out;

  const size_t NW = (size_t)DMODEL * DMODEL;   // 1M weight elems
  const size_t NX = (size_t)MROWS * DMODEL;    // 4M activation elems

  u16* ws  = (u16*)d_ws;
  u16* Wqb = ws;                 // bf16 weights
  u16* Wkb = Wqb + NW;
  u16* Wvb = Wkb + NW;
  u16* Wob = Wvb + NW;
  u16* Xq  = Wob + NW;           // bf16 inputs
  u16* Xk  = Xq + NX;
  u16* Xv  = Xk + NX;
  u16* Qb  = Xv + NX;            // bf16 projections
  u16* Kb  = Qb + NX;
  u16* Vb  = Kb + NX;
  u16* Cb  = Vb + NX;            // bf16 combined attention output
  // total: 4*1M + 7*4M = 32M u16 = 64 MB of workspace

  const int CT = 256;
  cvt_f32_bf16<<<(int)((NX + CT - 1) / CT), CT, 0, stream>>>(query, Xq, (int)NX);
  cvt_f32_bf16<<<(int)((NX + CT - 1) / CT), CT, 0, stream>>>(key,   Xk, (int)NX);
  cvt_f32_bf16<<<(int)((NX + CT - 1) / CT), CT, 0, stream>>>(value, Xv, (int)NX);
  cvt_f32_bf16<<<(int)((NW + CT - 1) / CT), CT, 0, stream>>>(Wq, Wqb, (int)NW);
  cvt_f32_bf16<<<(int)((NW + CT - 1) / CT), CT, 0, stream>>>(Wk, Wkb, (int)NW);
  cvt_f32_bf16<<<(int)((NW + CT - 1) / CT), CT, 0, stream>>>(Wv, Wvb, (int)NW);
  cvt_f32_bf16<<<(int)((NW + CT - 1) / CT), CT, 0, stream>>>(Wo, Wob, (int)NW);

  dim3 ggrid(DMODEL / 256, MROWS / 64);        // (4, 64)
  gemm_xwt<false><<<ggrid, 256, 0, stream>>>(Xq, Wqb, bq, Qb,
                                             MROWS, DMODEL, DMODEL);
  gemm_xwt<false><<<ggrid, 256, 0, stream>>>(Xk, Wkb, bk, Kb,
                                             MROWS, DMODEL, DMODEL);
  gemm_xwt<false><<<ggrid, 256, 0, stream>>>(Xv, Wvb, bv, Vb,
                                             MROWS, DMODEL, DMODEL);

  dim3 agrid(2 * NHEADS, S_LEN / 128);         // (32, 16)
  attn_flash<<<agrid, 256, 0, stream>>>(Qb, Kb, Vb, Cb);

  gemm_xwt<true><<<ggrid, 256, 0, stream>>>(Cb, Wob, bo, (void*)out,
                                            MROWS, DMODEL, DMODEL);
}
